// Dense_52432960749782
// MI455X (gfx1250) — 8-bit matrix operand order repaired and checked by reading, compiled, not run on hardware
//
#include <hip/hip_runtime.h>

typedef __attribute__((ext_vector_type(16))) int   v16i;
typedef __attribute__((ext_vector_type(8)))  float v8f;
typedef __attribute__((ext_vector_type(4)))  int   v4i;
typedef __attribute__((ext_vector_type(2)))  int   v2i;
typedef __attribute__((ext_vector_type(4)))  float v4f;

#define TOKENS 4096
#define DIN    1024
#define UNITS  4096

#define LDS_STRIDE 144
#define TILE_BYTES (128 * LDS_STRIDE)


__device__ __forceinline__ int pack4_fp8(float f0, float f1, float f2, float f3) {
    int p = 0;
    p = __builtin_amdgcn_cvt_pk_fp8_f32(f0, f1, p, false);
    p = __builtin_amdgcn_cvt_pk_fp8_f32(f2, f3, p, true);
    return p;
}

__global__ __launch_bounds__(256) void convert_x_fp8(const float* __restrict__ x,
                                                     int* __restrict__ apack) {
    int idx = blockIdx.x * blockDim.x + threadIdx.x;
    const v4f* src = (const v4f*)(x + (size_t)idx * 16);
    v4f f0 = src[0], f1 = src[1], f2 = src[2], f3 = src[3];
    v4i o;
    o.x = pack4_fp8(f0.x, f0.y, f0.z, f0.w);
    o.y = pack4_fp8(f1.x, f1.y, f1.z, f1.w);
    o.z = pack4_fp8(f2.x, f2.y, f2.z, f2.w);
    o.w = pack4_fp8(f3.x, f3.y, f3.z, f3.w);
    *(v4i*)(apack + (size_t)idx * 4) = o;
}

__global__ __launch_bounds__(256) void convert_w_fp8_t(const float* __restrict__ w,
                                                       char* __restrict__ bt) {
    int idx = blockIdx.x * blockDim.x + threadIdx.x;
    int n  = idx & (UNITS - 1);
    int k0 = (idx >> 12) * 16;
    const float* src = w + (size_t)k0 * UNITS + n;
    float f[16];
#pragma unroll
    for (int i = 0; i < 16; ++i) f[i] = src[(size_t)i * UNITS];
    v4i o;
    o.x = pack4_fp8(f[0],  f[1],  f[2],  f[3]);
    o.y = pack4_fp8(f[4],  f[5],  f[6],  f[7]);
    o.z = pack4_fp8(f[8],  f[9],  f[10], f[11]);
    o.w = pack4_fp8(f[12], f[13], f[14], f[15]);
    *(v4i*)(bt + (size_t)n * DIN + k0) = o;
}


__device__ __forceinline__ void async_copy_b128(unsigned lds_off, const char* gptr) {
    asm volatile("global_load_async_to_lds_b128 %0, %1, off"
                 :: "v"(lds_off), "v"(gptr) : "memory");
}

__device__ __forceinline__ void wait_async0() {
    asm volatile("s_wait_asynccnt 0x0" ::: "memory");
}

__device__ __forceinline__ float fast_tanh(float x) {
    float r;
    asm("v_tanh_f32 %0, %1" : "=v"(r) : "v"(x));
    return r;
}

__global__ __launch_bounds__(256) void gemm_fp8_gelu(
    const char* __restrict__ Apack,
    const char* __restrict__ Bt,
    const float* __restrict__ bias,
    float* __restrict__ out)
{
    __shared__ __attribute__((aligned(16))) char smem[4 * TILE_BYTES];

    const int t    = threadIdx.x;
    const int lane = t & 31;
    const int wave = t >> 5;
    const int g    = lane >> 4;
    const int r    = lane & 15;
    const int wm   = (wave >> 1) * 32;
    const int wn   = (wave & 1) * 64;
    const int m0   = blockIdx.y * 128 + wm;
    const int n0   = blockIdx.x * 128 + wn;

    const char* gA = Apack + (size_t)(blockIdx.y * 128) * DIN;
    const char* gB = Bt    + (size_t)(blockIdx.x * 128) * DIN;

    auto stage = [&](int s, int kb) {
        char* ldsA = smem + s * (2 * TILE_BYTES);
        char* ldsB = ldsA + TILE_BYTES;
#pragma unroll
        for (int c = 0; c < 4; ++c) {
            int lin = c * 4096 + t * 16;
            int row = lin >> 7;
            int ko  = lin & 127;
            async_copy_b128((unsigned)(size_t)(ldsA + row * LDS_STRIDE + ko),
                            gA + (size_t)row * DIN + kb + ko);
            async_copy_b128((unsigned)(size_t)(ldsB + row * LDS_STRIDE + ko),
                            gB + (size_t)row * DIN + kb + ko);
        }
    };

    v8f acc[2][4] = {};

    stage(0, 0);

    for (int i = 0; i < DIN / 128; ++i) {
        const int cur = i & 1;
        wait_async0();
        __syncthreads();

        if (i + 1 < DIN / 128)
            stage(1 - cur, (i + 1) * 128);

        const char* ldsA = smem + cur * (2 * TILE_BYTES);
        const char* ldsB = ldsA + TILE_BYTES;

        v16i a[2], b[4];

#pragma unroll
        for (int mt = 0; mt < 2; ++mt) {
            const char* p = ldsA + (size_t)(wm + mt * 16 + r) * LDS_STRIDE + g * 16;
#pragma unroll
            for (int j = 0; j < 4; ++j) {
                v4i q = *(const v4i*)(p + j * 32);
                a[mt][4 * j + 0] = q.x;
                a[mt][4 * j + 1] = q.y;
                a[mt][4 * j + 2] = q.z;
                a[mt][4 * j + 3] = q.w;
            }
        }

#pragma unroll
        for (int nt = 0; nt < 4; ++nt) {
            const char* p = ldsB + (size_t)(wn + nt * 16 + r) * LDS_STRIDE + g * 16;
#pragma unroll
            for (int j = 0; j < 4; ++j) {
                v4i q = *(const v4i*)(p + j * 32);
                b[nt][4 * j + 0] = q.x;
                b[nt][4 * j + 1] = q.y;
                b[nt][4 * j + 2] = q.z;
                b[nt][4 * j + 3] = q.w;
            }
        }

#pragma unroll
        for (int mt = 0; mt < 2; ++mt)
#pragma unroll
            for (int nt = 0; nt < 4; ++nt)
                acc[mt][nt] = __builtin_amdgcn_wmma_f32_16x16x128_fp8_fp8(
                    a[mt], b[nt], (short)0, acc[mt][nt], false, false);
    }

#pragma unroll
    for (int nt = 0; nt < 4; ++nt) {
        const int col = n0 + nt * 16 + r;
        const float bv = bias[col];
#pragma unroll
        for (int mt = 0; mt < 2; ++mt) {
#pragma unroll
            for (int rr = 0; rr < 8; ++rr) {
                const int row = m0 + mt * 16 + g * 8 + rr;
                float v = acc[mt][nt][rr] + bv;
                float u = fast_tanh(0.7978845608028654f * (v + 0.044715f * v * v * v));
                out[(size_t)row * UNITS + col] = 0.5f * v * (1.0f + u);
            }
        }
    }
}

extern "C" void kernel_launch(void* const* d_in, const int* in_sizes, int n_in,
                              void* d_out, int out_size, void* d_ws, size_t ws_size,
                              hipStream_t stream) {
    const float* x    = (const float*)d_in[0];
    const float* w    = (const float*)d_in[1];
    const float* bias = (const float*)d_in[2];
    float* out        = (float*)d_out;

    char* ws    = (char*)d_ws;
    int*  apack = (int*)ws;
    char* btpk  = ws + (size_t)TOKENS * DIN;

    convert_x_fp8<<<(TOKENS * DIN / 16) / 256, 256, 0, stream>>>(x, apack);
    convert_w_fp8_t<<<(UNITS * (DIN / 16)) / 256, 256, 0, stream>>>(w, btpk);

    dim3 grid(UNITS / 128, TOKENS / 128);
    gemm_fp8_gelu<<<grid, 256, 0, stream>>>((const char*)apack, (const char*)btpk,
                                            bias, out);
}
